// VAE_6365141533232
// MI455X (gfx1250) — compile-verified, matrix operand order checked by reading, not run on hardware
//
#include <hip/hip_runtime.h>
#include <hip/hip_bf16.h>

// ---------------------------------------------------------------------------
// VAE-LSTM for MI455X (gfx1250): wave32 + v_wmma_f32_16x16x32_f16.
// Each workgroup: 64 batch rows, 8 waves, loops over 27 timesteps.
// LSTM cell state kept in VGPRs (each wave owns 2 of 16 H column tiles).
// Weights pre-packed to fp16 B-fragment layout in workspace by prep kernels.
// x_t / eps_t tiles pipelined via GLOBAL_LOAD_ASYNC_TO_LDS (ASYNCcnt).
// ---------------------------------------------------------------------------

typedef _Float16 half_t;
typedef half_t v16h __attribute__((ext_vector_type(16)));
typedef float  v8f  __attribute__((ext_vector_type(8)));

#define B_TOT   65536
#define T_TOT   28
#define NSTEP   27
#define HDIM    256
#define ZDIM    10
#define ROWS    64          // batch rows per workgroup
#define NTHR    256         // 8 waves

// strides (in halfs) padded for LDS bank spread
#define AS_A    520         // LSTM A buffer [64 x 512]
#define AS_1    264         // scratch 1 (<=256 cols)
#define AS_2    136         // scratch 2 (<=128 cols)
#define FS      40          // f32 scratch stride

// workspace offsets in halfs (fragment buffers, each frag = 512 halfs = 1KB)
#define OFF_E1   0u            // 1x4   -> 2048
#define OFF_E2   2048u         // 2x8   -> 8192
#define OFF_E3   10240u        // 4x16  -> 32768
#define OFF_LSTM 43008u        // 4 gates x 16x16 -> 524288
#define OFF_MV   567296u       // 8x2   -> 8192
#define OFF_PJ   575488u       // 1x16  -> 8192
#define OFF_D1   583680u       // 8x8   -> 32768
#define OFF_D2   616448u       // 4x4   -> 8192
#define OFF_D3   624640u       // 2x2   -> 2048
#define OFF_END  626688u       // total halfs

union AFrag { v16h v; uint4 q[2]; unsigned int u[8]; };

__device__ inline v8f vzero8() {
  v8f z;
  #pragma unroll
  for (int r = 0; r < 8; ++r) z[r] = 0.f;
  return z;
}

// A fragment (16x32 f16) from LDS row-major.  lanes 0-15: M=lane, K pairs
// 0..7 & 16..23; lanes 16-31: M=lane-16, K pairs 8..15 & 24..31.
// => two contiguous 16B reads per lane (ds_load_b128 x2).
__device__ inline v16h load_a(const half_t* __restrict__ A, int stride, int mt,
                              int kbase, int lane) {
  int m  = mt * 16 + (lane & 15);
  int k0 = kbase + ((lane >> 4) << 3);
  const uint4* p = reinterpret_cast<const uint4*>(A + m * stride + k0);
  AFrag f;
  f.q[0] = p[0];     // K = k0+0..7
  f.q[1] = p[2];     // K = k0+16..23
  return f.v;
}

// B fragment from pre-swizzled global: lane-major, 16 halfs per lane.
__device__ inline v16h load_b(const half_t* __restrict__ W, int lane) {
  const uint4* p = reinterpret_cast<const uint4*>(W + lane * 16);
  AFrag f;
  f.q[0] = p[0];
  f.q[1] = p[1];
  return f.v;
}

__device__ inline float fast_sig(float x) { return 1.f / (1.f + __expf(-x)); }
__device__ inline float fast_tanh(float x) {
  float e = __expf(-2.f * fabsf(x));
  float t = (1.f - e) / (1.f + e);
  return (x >= 0.f) ? t : -t;
}

// generic tiled GEMM: D(f16, lds) = act( A(f16, lds) @ Wfrag + bias )
// Each wave owns fixed mt = wave&3; A-fragments cached in registers over all
// of the wave's N-tiles (nt = (wave>>2), step 2).
template <int Kt, int Nt, bool RELU>
__device__ inline void gemm_h(const half_t* __restrict__ Wf,
                              const float* __restrict__ bias, int Nvalid,
                              const half_t* __restrict__ A, int As,
                              half_t* __restrict__ D, int Ds, int tid) {
  const int lane = tid & 31, wave = tid >> 5;
  const int mt = wave & 3, ng = wave >> 2;
  v16h af[Kt];
  #pragma unroll
  for (int kt = 0; kt < Kt; ++kt) af[kt] = load_a(A, As, mt, kt * 32, lane);
  const int m0 = mt * 16 + ((lane >> 4) << 3);
  const int nl = lane & 15;
  for (int nt = ng; nt < Nt; nt += 2) {
    v8f acc = vzero8();
    const half_t* wp = Wf + (size_t)(nt * Kt) * 512;
    #pragma unroll
    for (int kt = 0; kt < Kt; ++kt) {
      v16h b = load_b(wp + kt * 512, lane);
      acc = __builtin_amdgcn_wmma_f32_16x16x32_f16(false, af[kt], false, b,
                                                   (short)0, acc, false, false);
    }
    const int n = nt * 16 + nl;
    const float bv = (n < Nvalid) ? bias[n] : 0.f;
    #pragma unroll
    for (int r = 0; r < 8; ++r) {
      float v = acc[r] + bv;
      if (RELU) v = fmaxf(v, 0.f);
      D[(m0 + r) * Ds + n] = (half_t)v;
    }
  }
}

// GEMM with f32 LDS output (mean/var head, decoder final layer)
template <int Kt, int Nt>
__device__ inline void gemm_f(const half_t* __restrict__ Wf,
                              const float* __restrict__ bias, int Nvalid,
                              const half_t* __restrict__ A, int As,
                              float* __restrict__ D, int Ds, int tid) {
  const int lane = tid & 31, wave = tid >> 5;
  const int mt = wave & 3, ng = wave >> 2;
  v16h af[Kt];
  #pragma unroll
  for (int kt = 0; kt < Kt; ++kt) af[kt] = load_a(A, As, mt, kt * 32, lane);
  const int m0 = mt * 16 + ((lane >> 4) << 3);
  const int nl = lane & 15;
  for (int nt = ng; nt < Nt; nt += 2) {
    v8f acc = vzero8();
    const half_t* wp = Wf + (size_t)(nt * Kt) * 512;
    #pragma unroll
    for (int kt = 0; kt < Kt; ++kt) {
      v16h b = load_b(wp + kt * 512, lane);
      acc = __builtin_amdgcn_wmma_f32_16x16x32_f16(false, af[kt], false, b,
                                                   (short)0, acc, false, false);
    }
    const int n = nt * 16 + nl;
    const float bv = (n < Nvalid) ? bias[n] : 0.f;
    #pragma unroll
    for (int r = 0; r < 8; ++r)
      D[(m0 + r) * Ds + n] = acc[r] + bv;
  }
}

// ---------------------------------------------------------------------------
// main persistent kernel
// ---------------------------------------------------------------------------
__global__ __launch_bounds__(NTHR, 1)
void vae_main(const float* __restrict__ x, const float* __restrict__ eps,
              float* __restrict__ out,
              const half_t* __restrict__ wsH,
              const float* __restrict__ gateB, const float* __restrict__ mvB,
              const float* __restrict__ eb1, const float* __restrict__ eb2,
              const float* __restrict__ eb3, const float* __restrict__ pb,
              const float* __restrict__ db1, const float* __restrict__ db2,
              const float* __restrict__ db3) {
  __shared__ __align__(16) half_t Abuf[ROWS * AS_A];   // [e | h] for LSTM
  __shared__ __align__(16) half_t SB1[ROWS * AS_1];
  __shared__ __align__(16) half_t SB2[ROWS * AS_2];
  __shared__ __align__(16) float  FB[ROWS * FS];
  __shared__ __align__(16) float  XST[ROWS * 32];      // async x_t staging
  __shared__ __align__(16) float  EST[ROWS * 16];      // async eps_t staging

  const int tid = threadIdx.x, lane = tid & 31, wave = tid >> 5;
  const int rb = blockIdx.x * ROWS;

  const unsigned xst_base = (unsigned)(size_t)(&XST[0]);
  const unsigned est_base = (unsigned)(size_t)(&EST[0]);

  // async load of x tile for step t into XST (GVS: sgpr base + vgpr offset)
  auto async_x = [&](int t) {
    #pragma unroll
    for (int k = 0; k < 8; ++k) {
      int i = tid + k * NTHR;            // 0..2047
      int r = i >> 5, c = i & 31;
      if (c < 28) {
        unsigned loff = xst_base + (unsigned)(i * 4);
        unsigned goff =
            ((unsigned)(rb + r) * (unsigned)(T_TOT * 28) +
             (unsigned)t * 28u + (unsigned)c) * 4u;
        asm volatile("global_load_async_to_lds_b32 %0, %1, %2"
                     :: "v"(loff), "v"(goff), "s"(x) : "memory");
      }
    }
  };
  // async load of eps tile for step t into EST
  auto async_eps = [&](int t) {
    #pragma unroll
    for (int k = 0; k < 4; ++k) {
      int i = tid + k * NTHR;            // 0..1023
      int r = i >> 4, c = i & 15;
      if (c < ZDIM) {
        unsigned loff = est_base + (unsigned)(i * 4);
        unsigned goff =
            (((unsigned)t * (unsigned)B_TOT + (unsigned)(rb + r)) *
             (unsigned)ZDIM + (unsigned)c) * 4u;
        asm volatile("global_load_async_to_lds_b32 %0, %1, %2"
                     :: "v"(loff), "v"(goff), "s"(eps) : "memory");
      }
    }
  };

  // prologue: kick off t=0 tiles, zero A buffer (h_prev = 0)
  async_x(0);
  async_eps(0);
  for (int i = tid; i < ROWS * AS_A; i += NTHR) Abuf[i] = (half_t)0.f;

  const half_t* WE1 = wsH + OFF_E1;
  const half_t* WE2 = wsH + OFF_E2;
  const half_t* WE3 = wsH + OFF_E3;
  const half_t* WL  = wsH + OFF_LSTM;
  const half_t* WMV = wsH + OFF_MV;
  const half_t* WPJ = wsH + OFF_PJ;
  const half_t* WD1 = wsH + OFF_D1;
  const half_t* WD2 = wsH + OFF_D2;
  const half_t* WD3 = wsH + OFF_D3;

  // persistent cell state: wave owns H tiles nh = wave*2, wave*2+1
  v8f cst[2][4];
  #pragma unroll
  for (int nn = 0; nn < 2; ++nn)
    #pragma unroll
    for (int mt = 0; mt < 4; ++mt) cst[nn][mt] = vzero8();

  float lossacc = 0.f;

  for (int t = 0; t < NSTEP; ++t) {
    // ---- wait for this step's async tiles (x_t, eps_t), all waves ----
    asm volatile("s_wait_asynccnt 0" ::: "memory");
    __syncthreads();

    // ---- convert staged x_t into SB2 [64 x 32] fp16 (pad 28->32) ----
    for (int i = tid; i < ROWS * 32; i += NTHR) {
      int r = i >> 5, c = i & 31;
      float v = (c < 28) ? XST[i] : 0.f;
      SB2[r * AS_2 + c] = (half_t)v;
    }
    // prefetch reconstruction-target row (t+1) for the loss phase
    if (tid < ROWS)
      __builtin_prefetch(&x[(size_t)(rb + tid) * (T_TOT * 28) +
                            (size_t)(t + 1) * 28], 0, 1);
    __syncthreads();

    // XST consumed: launch next step's x tile
    if (t + 1 < NSTEP) async_x(t + 1);

    // ---- encoder ----
    gemm_h<1, 4,  true>(WE1, eb1, 64,  SB2, AS_2, SB1, AS_1, tid);  __syncthreads();
    gemm_h<2, 8,  true>(WE2, eb2, 128, SB1, AS_1, SB2, AS_2, tid);  __syncthreads();
    gemm_h<4, 16, true>(WE3, eb3, 256, SB2, AS_2, Abuf, AS_A, tid); __syncthreads();

    // ---- LSTM: gates = [e|h] @ [Wx;Wh], cell state in VGPRs ----
    for (int nn = 0; nn < 2; ++nn) {
      const int nh = wave * 2 + nn;
      v8f acc[4][4];            // [gate][mt]
      #pragma unroll
      for (int g = 0; g < 4; ++g)
        #pragma unroll
        for (int mt = 0; mt < 4; ++mt) acc[g][mt] = vzero8();

      const half_t* w0 = WL + (size_t)0 * 131072 + (size_t)nh * 8192;
      const half_t* w1 = WL + (size_t)1 * 131072 + (size_t)nh * 8192;
      const half_t* w2 = WL + (size_t)2 * 131072 + (size_t)nh * 8192;
      const half_t* w3 = WL + (size_t)3 * 131072 + (size_t)nh * 8192;

      #pragma unroll 4
      for (int kt = 0; kt < 16; ++kt) {
        v16h b0 = load_b(w0 + kt * 512, lane);
        v16h b1 = load_b(w1 + kt * 512, lane);
        v16h b2 = load_b(w2 + kt * 512, lane);
        v16h b3 = load_b(w3 + kt * 512, lane);
        #pragma unroll
        for (int mt = 0; mt < 4; ++mt) {
          v16h a = load_a(Abuf, AS_A, mt, kt * 32, lane);
          acc[0][mt] = __builtin_amdgcn_wmma_f32_16x16x32_f16(false, a, false, b0, (short)0, acc[0][mt], false, false);
          acc[1][mt] = __builtin_amdgcn_wmma_f32_16x16x32_f16(false, a, false, b1, (short)0, acc[1][mt], false, false);
          acc[2][mt] = __builtin_amdgcn_wmma_f32_16x16x32_f16(false, a, false, b2, (short)0, acc[2][mt], false, false);
          acc[3][mt] = __builtin_amdgcn_wmma_f32_16x16x32_f16(false, a, false, b3, (short)0, acc[3][mt], false, false);
        }
      }

      const int n = nh * 16 + (lane & 15);
      const float bi = gateB[n], bf = gateB[256 + n],
                  bo = gateB[512 + n], bc = gateB[768 + n];
      const int m0base = ((lane >> 4) << 3);
      #pragma unroll
      for (int mt = 0; mt < 4; ++mt) {
        v8f co = cst[nn][mt];
        #pragma unroll
        for (int r = 0; r < 8; ++r) {
          float gi = fast_sig(acc[0][mt][r] + bi);
          float gf = fast_sig(acc[1][mt][r] + bf);
          float go = fast_sig(acc[2][mt][r] + bo);
          float gc = fast_tanh(acc[3][mt][r] + bc);
          float cn = gf * co[r] + gi * gc;
          co[r] = cn;
          float hn = go * fast_tanh(cn);
          SB1[(mt * 16 + m0base + r) * AS_1 + n] = (half_t)hn;  // stage h_new
        }
        cst[nn][mt] = co;
      }
    }
    __syncthreads();

    // ---- copy staged h into A buffer cols [256..512) + mean/var GEMM ----
    for (int i = tid; i < ROWS * 128; i += NTHR) {
      int r = i >> 7, c2 = (i & 127) << 1;
      *reinterpret_cast<unsigned int*>(&Abuf[r * AS_A + 256 + c2]) =
          *reinterpret_cast<const unsigned int*>(&SB1[r * AS_1 + c2]);
    }
    gemm_f<8, 2>(WMV, mvB, 32, SB1, AS_1, FB, FS, tid);
    __syncthreads();

    // ---- reparameterize: z = eps*exp(0.5*logvar)+mean, KL accumulate ----
    for (int i = tid; i < ROWS * 32; i += NTHR) {
      int r = i >> 5, c = i & 31;
      float zv = 0.f;
      if (c < ZDIM) {
        float mn = FB[r * FS + c];
        float lv = FB[r * FS + 16 + c];
        float ep = EST[r * 16 + c];
        zv = ep * __expf(0.5f * lv) + mn;
        lossacc += -0.5f * (1.f + lv - mn * mn - __expf(lv));
      }
      SB2[r * AS_2 + c] = (half_t)zv;
    }
    __syncthreads();

    // EST consumed: launch next step's eps tile
    if (t + 1 < NSTEP) async_eps(t + 1);

    // ---- proj + decoder ----
    gemm_h<1, 16, false>(WPJ, pb,  256, SB2, AS_2, SB1, AS_1, tid); __syncthreads();
    gemm_h<8, 8,  true >(WD1, db1, 128, SB1, AS_1, SB2, AS_2, tid); __syncthreads();
    gemm_h<4, 4,  true >(WD2, db2, 64,  SB2, AS_2, SB1, AS_1, tid); __syncthreads();
    gemm_f<2, 2>(WD3, db3, 28, SB1, AS_1, FB, FS, tid);             __syncthreads();

    // ---- write y_hat, reconstruction loss (no integer division) ----
    {
      const int r = tid >> 2, c0 = tid & 3;       // 64 rows x 4 col-phases
      const float* fb = &FB[r * FS];
      const size_t obase =
          (size_t)(rb + r) * (T_TOT * 28) + (size_t)(t + 1) * 28;
      #pragma unroll
      for (int k = 0; k < 7; ++k) {
        int c = c0 + 4 * k;                       // covers 0..27
        float yv = fb[c];
        out[obase + c] = yv;
        float d = yv - x[obase + c];
        lossacc += d * d;
      }
    }
    __syncthreads();
  }

  // ---- block loss reduction + global atomic ----
  FB[tid] = lossacc;
  __syncthreads();
  for (int s = 128; s > 0; s >>= 1) {
    if (tid < s) FB[tid] += FB[tid + s];
    __syncthreads();
  }
  if (tid == 0)
    atomicAdd(out + (size_t)B_TOT * (T_TOT * 28),
              FB[0] * (1.f / ((float)B_TOT * (float)NSTEP)));
}

// ---------------------------------------------------------------------------
// prep kernels: pack fp32 weights -> fp16 B-fragment layout in workspace
// frag element (lane, ii): n = nt*16 + (lane&15),
// k = kt*32 + (lane>=16 ? 8:0) + (j<4 ? 2j : 16+2(j-4)) + (ii&1), j=ii>>1
// ---------------------------------------------------------------------------
__device__ inline int klocal_of(int lane, int ii) {
  int j = ii >> 1, e = ii & 1;
  return ((lane & 16) ? 8 : 0) + ((j < 4) ? 2 * j : 16 + 2 * (j - 4)) + e;
}

__global__ void pack_generic(const float* __restrict__ src, int K, int N,
                             int Ktp, int Ntp, half_t* __restrict__ dst) {
  const int total = Ktp * Ntp * 512;
  for (int idx = blockIdx.x * blockDim.x + threadIdx.x; idx < total;
       idx += gridDim.x * blockDim.x) {
    int frag = idx >> 9, r = idx & 511;
    int lane = r >> 4, ii = r & 15;
    int kt = frag % Ktp, nt = frag / Ktp;
    int k = kt * 32 + klocal_of(lane, ii);
    int n = nt * 16 + (lane & 15);
    float v = (k < K && n < N) ? src[(size_t)k * N + n] : 0.f;
    dst[idx] = (half_t)v;
  }
}

__global__ void pack_lstm(const float* __restrict__ Wx, const float* __restrict__ Wh,
                          const float* __restrict__ bx, const float* __restrict__ bh,
                          half_t* __restrict__ dst, float* __restrict__ dstB) {
  const int WTOT = 4 * 16 * 16 * 512;
  for (int idx = blockIdx.x * blockDim.x + threadIdx.x; idx < WTOT + 1024;
       idx += gridDim.x * blockDim.x) {
    if (idx < WTOT) {
      int g = idx / 131072, rem = idx % 131072;
      int frag = rem >> 9, r = rem & 511;
      int kt = frag & 15, nt = frag >> 4;
      int lane = r >> 4, ii = r & 15;
      int k = kt * 32 + klocal_of(lane, ii);          // 0..511
      int n = nt * 16 + (lane & 15);
      float v = (k < 256)
                    ? Wx[((size_t)g * 256 + k) * 256 + n]
                    : Wh[((size_t)g * 256 + (k - 256)) * 256 + n];
      dst[idx] = (half_t)v;
    } else {
      int b = idx - WTOT;
      dstB[b] = bx[b] + bh[b];
    }
  }
}

__global__ void pack_mv(const float* __restrict__ mW, const float* __restrict__ vW,
                        const float* __restrict__ mb, const float* __restrict__ vb,
                        half_t* __restrict__ dst, float* __restrict__ dstB) {
  const int WTOT = 8 * 2 * 512;
  for (int idx = blockIdx.x * blockDim.x + threadIdx.x; idx < WTOT + 32;
       idx += gridDim.x * blockDim.x) {
    if (idx < WTOT) {
      int frag = idx >> 9, r = idx & 511;
      int kt = frag % 8, nt = frag / 8;
      int lane = r >> 4, ii = r & 15;
      int k = kt * 32 + klocal_of(lane, ii);
      int n = nt * 16 + (lane & 15);                  // 0..31
      float v = 0.f;
      if (k < 256) {
        if (n < 10)                 v = mW[(size_t)k * 10 + n];
        else if (n >= 16 && n < 26) v = vW[(size_t)k * 10 + (n - 16)];
      }
      dst[idx] = (half_t)v;
    } else {
      int n = idx - WTOT;
      float v = 0.f;
      if (n < 10)                 v = mb[n];
      else if (n >= 16 && n < 26) v = vb[n - 16];
      dstB[n] = v;
    }
  }
}

__global__ void zero_out_k(float* __restrict__ out) {
  int idx = blockIdx.x * blockDim.x + threadIdx.x;
  const int TOT = B_TOT * 28;
  if (idx < TOT) out[(size_t)(idx / 28) * (T_TOT * 28) + (idx % 28)] = 0.f;
  if (idx == 0) out[(size_t)B_TOT * (T_TOT * 28)] = 0.f;
}

// ---------------------------------------------------------------------------
extern "C" void kernel_launch(void* const* d_in, const int* in_sizes, int n_in,
                              void* d_out, int out_size, void* d_ws, size_t ws_size,
                              hipStream_t stream) {
  const float* x    = (const float*)d_in[0];
  const float* eps  = (const float*)d_in[1];
  const float* eW1  = (const float*)d_in[2];  const float* eb1 = (const float*)d_in[3];
  const float* eW2  = (const float*)d_in[4];  const float* eb2 = (const float*)d_in[5];
  const float* eW3  = (const float*)d_in[6];  const float* eb3 = (const float*)d_in[7];
  const float* lWx  = (const float*)d_in[8];  const float* lbx = (const float*)d_in[9];
  const float* lWh  = (const float*)d_in[10]; const float* lbh = (const float*)d_in[11];
  const float* mW   = (const float*)d_in[12]; const float* mb  = (const float*)d_in[13];
  const float* vW   = (const float*)d_in[14]; const float* vb  = (const float*)d_in[15];
  const float* pW   = (const float*)d_in[16]; const float* pb  = (const float*)d_in[17];
  const float* dW1  = (const float*)d_in[18]; const float* db1 = (const float*)d_in[19];
  const float* dW2  = (const float*)d_in[20]; const float* db2 = (const float*)d_in[21];
  const float* dW3  = (const float*)d_in[22]; const float* db3 = (const float*)d_in[23];
  float* out = (float*)d_out;

  half_t* wsH  = (half_t*)d_ws;
  float* gateB = (float*)((char*)d_ws + (size_t)OFF_END * 2);
  float* mvB   = gateB + 1024;

  zero_out_k<<<(B_TOT * 28 + 255) / 256, 256, 0, stream>>>(out);

  pack_generic<<<8,   256, 0, stream>>>(eW1, 28,  64,  1, 4,  wsH + OFF_E1);
  pack_generic<<<32,  256, 0, stream>>>(eW2, 64,  128, 2, 8,  wsH + OFF_E2);
  pack_generic<<<128, 256, 0, stream>>>(eW3, 128, 256, 4, 16, wsH + OFF_E3);
  pack_generic<<<32,  256, 0, stream>>>(pW,  10,  256, 1, 16, wsH + OFF_PJ);
  pack_generic<<<128, 256, 0, stream>>>(dW1, 256, 128, 8, 8,  wsH + OFF_D1);
  pack_generic<<<32,  256, 0, stream>>>(dW2, 128, 64,  4, 4,  wsH + OFF_D2);
  pack_generic<<<8,   256, 0, stream>>>(dW3, 64,  28,  2, 2,  wsH + OFF_D3);
  pack_lstm<<<2052, 256, 0, stream>>>(lWx, lWh, lbx, lbh, wsH + OFF_LSTM, gateB);
  pack_mv<<<33, 256, 0, stream>>>(mW, vW, mb, vb, wsH + OFF_MV, mvB);

  vae_main<<<B_TOT / ROWS, NTHR, 0, stream>>>(
      x, eps, out, wsH, gateB, mvB, eb1, eb2, eb3, pb, db1, db2, db3);
}
